// LieAlgebraConnection_49185965474163
// MI455X (gfx1250) — compile-verified, matrix operand order checked by reading, not run on hardware
//
#include <hip/hip_runtime.h>
#include <hip/hip_bf16.h>

// ---------------------------------------------------------------------------
// LieAlgebraConnection: out = x @ expm(W - W^T)
//   x: [4,4096,1024] fp32 (treated [16384,1024]);  W: [1024,1024] fp32
//
// expm via scaling-and-squaring: B = (W - W^T)/2^6 (||B||~0.18), 10 Horner
// terms, 6 squarings. Every matmul is a bf16x3-split WMMA GEMM
// (v_wmma_f32_16x16x32_bf16, fp32 accumulate).
//
// Operands are pre-split once into bf16 hi/lo planes (normal layout for the
// A operand, transposed layout for the B operand). GEMM tile staging uses
// double-buffered GLOBAL_LOAD_ASYNC_TO_LDS_B128 (ASYNCcnt) so global latency
// overlaps the WMMA stream; the inner loop has no conversion VALU.
// ---------------------------------------------------------------------------

typedef __attribute__((ext_vector_type(16))) __bf16 v16bf;
typedef __attribute__((ext_vector_type(8)))  __bf16 bf8v;
typedef __attribute__((ext_vector_type(8)))  float  v8f;

#define DIMN 1024
#define BM 128
#define BN 128
#define BK 32
#define LDS_STRIDE 40   // 32 K elems + pad (multiple of 8 keeps 16B alignment)

static __device__ __forceinline__ __bf16 f2bf_rne(float f) {
    unsigned u = __builtin_bit_cast(unsigned, f);
    unsigned r = u + 0x7FFFu + ((u >> 16) & 1u);   // round-to-nearest-even
    unsigned short h = (unsigned short)(r >> 16);
    return __builtin_bit_cast(__bf16, h);
}
static __device__ __forceinline__ float bf2f(__bf16 b) {
    unsigned short h = __builtin_bit_cast(unsigned short, b);
    unsigned u = ((unsigned)h) << 16;
    return __builtin_bit_cast(float, u);
}

// One 16x32 bf16 WMMA operand fragment from an LDS plane (ISA 7.12.2 layout:
// lanes 0-15 row M=lane hold K=[0..7,16..23]; lanes 16-31 K=[8..15,24..31]).
static __device__ __forceinline__ v16bf
ld_frag(const __bf16* plane, int base, int lane) {
    const int r    = base + (lane & 15);
    const int koff = (lane >> 4) << 3;          // 0 or 8
    const __bf16* p = plane + r * LDS_STRIDE;
    bf8v r0 = *(const bf8v*)(p + koff);
    bf8v r1 = *(const bf8v*)(p + koff + 16);
    return __builtin_shufflevector(r0, r1,
        0, 1, 2, 3, 4, 5, 6, 7, 8, 9, 10, 11, 12, 13, 14, 15);
}

// C[M,N] = alpha * (A[M,K] * B[K,N]) (+ I), operands pre-split:
//   Ah/Al: bf16 planes of A, row-major [M,K]
//   BTh/BTl: bf16 planes of B^T, row-major [N,K]  (B-operand columns)
// M%128==0, N%128==0, K%64==0. 256 threads = 8 waves (wave32).
__global__ __launch_bounds__(256)
void gemm_bf16x3_async(const __bf16* __restrict__ Ah, const __bf16* __restrict__ Al,
                       const __bf16* __restrict__ BTh, const __bf16* __restrict__ BTl,
                       float* __restrict__ C, int M, int N, int K,
                       float alpha, int addIdentity) {
    // double-buffered tiles: 2 bufs x 4 planes x [128 x LDS_STRIDE] bf16 = 80KB
    __shared__ __attribute__((aligned(16))) __bf16 lds[2][4][BM * LDS_STRIDE];

    const int tid    = threadIdx.x;
    const int lane   = tid & 31;
    const int waveId = tid >> 5;           // 0..7
    const int wm     = waveId & 3;         // 4 row groups of 32
    const int wn     = waveId >> 2;        // 2 col groups of 64

    const int blockM = blockIdx.y * BM;
    const int blockN = blockIdx.x * BN;

    const __bf16* gplane[4] = { Ah, Al, BTh, BTl };

    // Issue the async staging for one k-tile into LDS buffer bufIdx.
    // 16B chunk = 8 bf16; 4 chunks/row; 512 chunks/plane; 2 per thread.
    auto issue_tile = [&](int kt, int bufIdx) {
        const int k0 = kt * BK;
        #pragma unroll
        for (int p = 0; p < 4; ++p) {
            const int rbase = (p < 2) ? blockM : blockN;
            #pragma unroll
            for (int i = 0; i < 2; ++i) {
                int idx = tid + i * 256;          // 0..511
                int row = idx >> 2;               // 0..127
                int kc  = (idx & 3) << 3;         // 0,8,16,24
                const __bf16* g = gplane[p] + (size_t)(rbase + row) * K + k0 + kc;
                // low 32 bits of the generic shared-aperture address are the
                // wave-relative LDS byte address
                unsigned ldsAddr =
                    (unsigned)(size_t)(&lds[bufIdx][p][row * LDS_STRIDE + kc]);
                asm volatile(
                    "global_load_async_to_lds_b128 %0, %1, off"
                    :
                    : "v"(ldsAddr), "v"(g)
                    : "memory");
            }
        }
    };

    v8f acc[2][4];
    #pragma unroll
    for (int mi = 0; mi < 2; ++mi)
        #pragma unroll
        for (int ni = 0; ni < 4; ++ni) {
            v8f z = {};
            acc[mi][ni] = z;
        }

    const int nk = K / BK;
    issue_tile(0, 0);

    for (int kt = 0; kt < nk; ++kt) {
        const int cur = kt & 1;
        // current tile's async copies complete (next tile not yet issued)
        asm volatile("s_wait_asynccnt 0" ::: "memory");
        __syncthreads();   // cur buffer visible to all waves; prev reads done
        if (kt + 1 < nk) issue_tile(kt + 1, cur ^ 1);

        const __bf16* bufA_h = lds[cur][0];
        const __bf16* bufA_l = lds[cur][1];
        const __bf16* bufB_h = lds[cur][2];
        const __bf16* bufB_l = lds[cur][3];

        // ---- operand fragments ----
        v16bf aH[2], aL[2], bH[4], bL[4];
        #pragma unroll
        for (int mi = 0; mi < 2; ++mi) {
            aH[mi] = ld_frag(bufA_h, wm * 32 + mi * 16, lane);
            aL[mi] = ld_frag(bufA_l, wm * 32 + mi * 16, lane);
        }
        #pragma unroll
        for (int ni = 0; ni < 4; ++ni) {
            bH[ni] = ld_frag(bufB_h, wn * 64 + ni * 16, lane);
            bL[ni] = ld_frag(bufB_l, wn * 64 + ni * 16, lane);
        }

        // ---- bf16x3: C += aH*bH + aH*bL + aL*bH ----
        #pragma unroll
        for (int mi = 0; mi < 2; ++mi) {
            #pragma unroll
            for (int ni = 0; ni < 4; ++ni) {
                acc[mi][ni] = __builtin_amdgcn_wmma_f32_16x16x32_bf16(
                    false, aH[mi], false, bH[ni], (short)0, acc[mi][ni], false, false);
                acc[mi][ni] = __builtin_amdgcn_wmma_f32_16x16x32_bf16(
                    false, aH[mi], false, bL[ni], (short)0, acc[mi][ni], false, false);
                acc[mi][ni] = __builtin_amdgcn_wmma_f32_16x16x32_bf16(
                    false, aL[mi], false, bH[ni], (short)0, acc[mi][ni], false, false);
            }
        }
    }

    // ---- epilogue: C = alpha*acc (+ I). C/D layout per ISA 7.12.2:
    // lanes 0-15: N=lane, M=i; lanes 16-31: N=lane-16, M=8+i.
    const int nBaseLane = blockN + wn * 64 + (lane & 15);
    const int mHalf     = (lane >> 4) * 8;
    #pragma unroll
    for (int mi = 0; mi < 2; ++mi) {
        #pragma unroll
        for (int ni = 0; ni < 4; ++ni) {
            const int n  = nBaseLane + ni * 16;
            const int m0 = blockM + wm * 32 + mi * 16 + mHalf;
            #pragma unroll
            for (int i = 0; i < 8; ++i) {
                float v = alpha * acc[mi][ni][i];
                if (addIdentity && (m0 + i) == n) v += 1.0f;
                C[(size_t)(m0 + i) * N + n] = v;
            }
        }
    }
}

// Split fp32 [R,Cc] into bf16 hi/lo planes; optionally also transposed planes.
__global__ __launch_bounds__(256)
void split_kernel(const float* __restrict__ src,
                  __bf16* __restrict__ hiN, __bf16* __restrict__ loN,
                  __bf16* __restrict__ hiT, __bf16* __restrict__ loT,
                  int R, int Cc, int wantT) {
    int idx = blockIdx.x * 256 + threadIdx.x;
    float f = src[idx];
    __bf16 h = f2bf_rne(f);
    __bf16 l = f2bf_rne(f - bf2f(h));
    hiN[idx] = h;
    loN[idx] = l;
    if (wantT) {
        int i = idx / Cc, j = idx - i * Cc;
        size_t t = (size_t)j * R + i;
        hiT[t] = h;
        loT[t] = l;
    }
}

// val = (W[i][j]-W[j][i])*scale + diag*(i==j), split into N and T planes.
__global__ __launch_bounds__(256)
void skew_split_kernel(const float* __restrict__ w,
                       __bf16* __restrict__ hiN, __bf16* __restrict__ loN,
                       __bf16* __restrict__ hiT, __bf16* __restrict__ loT,
                       float scale, float diag) {
    int idx = blockIdx.x * 256 + threadIdx.x;
    int i = idx >> 10;
    int j = idx & (DIMN - 1);
    float v = (w[(size_t)i * DIMN + j] - w[(size_t)j * DIMN + i]) * scale
              + ((i == j) ? diag : 0.0f);
    __bf16 h = f2bf_rne(v);
    __bf16 l = f2bf_rne(v - bf2f(h));
    hiN[idx] = h;
    loN[idx] = l;
    size_t t = (size_t)j * DIMN + i;
    hiT[t] = h;
    loT[t] = l;
}

extern "C" void kernel_launch(void* const* d_in, const int* in_sizes, int n_in,
                              void* d_out, int out_size, void* d_ws, size_t ws_size,
                              hipStream_t stream) {
    const float* x = (const float*)d_in[0];   // [16384, 1024]
    const float* w = (const float*)d_in[1];   // [1024, 1024]
    float* out = (float*)d_out;

    const int D = DIMN;
    const size_t MAT = (size_t)D * D;         // 1M elems
    const int Mbig = 4 * 4096;                // 16384
    const size_t XEL = (size_t)Mbig * D;      // 16M elems

    // ---- workspace layout ----
    float*  Qf  = (float*)d_ws;               // fp32 GEMM result   (4 MB)
    __bf16* BhN = (__bf16*)(Qf + MAT);        // 8 square planes, 2 MB each
    __bf16* BlN = BhN + MAT;
    __bf16* BhT = BlN + MAT;
    __bf16* BlT = BhT + MAT;
    __bf16* PhN = BlT + MAT;
    __bf16* PlN = PhN + MAT;
    __bf16* PhT = PlN + MAT;
    __bf16* PlT = PhT + MAT;
    __bf16* Xh  = PlT + MAT;                  // 32 MB
    __bf16* Xl  = Xh + XEL;                   // 32 MB

    dim3 blk(256);
    dim3 gridEw((unsigned)(MAT / 256));
    dim3 gridX((unsigned)(XEL / 256));
    dim3 gridSq(D / BN, D / BM);
    dim3 gridBig(D / BN, Mbig / BM);

    const int   NTERMS = 10;                  // Taylor truncation ~1e-16
    const float SCALE  = 1.0f / 64.0f;        // 2^-6 scaling, 6 squarings

    // B = (W - W^T)/64  -> B planes (N for A-operand use)
    skew_split_kernel<<<gridEw, blk, 0, stream>>>(w, BhN, BlN, BhT, BlT,
                                                  SCALE, 0.0f);
    // P0 = I + B/NTERMS -> P planes (T for B-operand use)
    skew_split_kernel<<<gridEw, blk, 0, stream>>>(w, PhN, PlN, PhT, PlT,
                                                  SCALE / (float)NTERMS, 1.0f);

    // Horner: P <- I + (B*P)/k for k = NTERMS-1 .. 1 (fused in epilogue)
    for (int k = NTERMS - 1; k >= 1; --k) {
        gemm_bf16x3_async<<<gridSq, blk, 0, stream>>>(
            BhN, BlN, PhT, PlT, Qf, D, D, D, 1.0f / (float)k, 1);
        split_kernel<<<gridEw, blk, 0, stream>>>(Qf, PhN, PlN, PhT, PlT, D, D, 1);
    }

    // 6 squarings: P <- P*P
    for (int s = 0; s < 6; ++s) {
        gemm_bf16x3_async<<<gridSq, blk, 0, stream>>>(
            PhN, PlN, PhT, PlT, Qf, D, D, D, 1.0f, 0);
        split_kernel<<<gridEw, blk, 0, stream>>>(Qf, PhN, PlN, PhT, PlT, D, D, 1);
    }

    // split x once (A operand only; no transpose needed)
    split_kernel<<<gridX, blk, 0, stream>>>(x, Xh, Xl, (__bf16*)0, (__bf16*)0,
                                            Mbig, D, 0);

    // out = x @ R
    gemm_bf16x3_async<<<gridBig, blk, 0, stream>>>(
        Xh, Xl, PhT, PlT, out, Mbig, D, D, 1.0f, 0);
}
